// LabelGenerator_7773890805956
// MI455X (gfx1250) — matrix operand order repaired and checked by reading, not compiled, not run on hardware
//
#include <hip/hip_runtime.h>

#define IMG_H 240
#define IMG_W 320
#define NB 32
#define NS 10
#define NBS 320          // NB*NS warped images
#define HW 76800         // IMG_H*IMG_W
#define TOPK 512
#define CAND_MAX 4096

typedef __attribute__((ext_vector_type(16))) _Float16 v16h;
typedef __attribute__((ext_vector_type(8)))  float    v8f;

// ---------------------------------------------------------------- 3x3 inverse
__global__ void invert3x3_kernel(const float* __restrict__ src,
                                 float* __restrict__ dst, int n) {
    int i = blockIdx.x * blockDim.x + threadIdx.x;
    if (i >= n) return;
    const float* m = src + i * 9;
    float a = m[0], b = m[1], c = m[2];
    float d = m[3], e = m[4], f = m[5];
    float g = m[6], h = m[7], k = m[8];
    float A =  (e * k - f * h);
    float Bm = -(d * k - f * g);
    float C =  (d * h - e * g);
    float det = a * A + b * Bm + c * C;
    float inv = 1.0f / det;
    float* o = dst + i * 9;
    o[0] =  A * inv;               o[1] = -(b * k - c * h) * inv; o[2] =  (b * f - c * e) * inv;
    o[3] =  Bm * inv;              o[4] =  (a * k - c * g) * inv; o[5] = -(a * f - c * d) * inv;
    o[6] =  C * inv;               o[7] = -(a * h - b * g) * inv; o[8] =  (a * e - b * d) * inv;
}

// ---------------- pack W[64,64] into WMMA B-fragment layout as f16 (run once)
// entry e = ((nt*2+chunk)*32 + lane)*8 + v  holds halves for K=k0,k0+1 at col n
__global__ void pack_b_kernel(const float* __restrict__ Wd,
                              _Float16* __restrict__ bp) {
    for (int e = threadIdx.x; e < 2048; e += blockDim.x) {
        int v    = e & 7;
        int slot = e >> 3;
        int lane = slot & 31;
        int g    = slot >> 5;       // 0..7
        int chunk = g & 1;
        int nt    = g >> 1;
        int half  = lane >> 4;
        int n  = nt * 16 + (lane & 15);
        int k0 = chunk * 32 + (v >> 2) * 16 + half * 8 + (v & 3) * 2;   // B 32x16 f16 layout (table order, matches A)
        bp[2 * e]     = (_Float16)Wd[k0 * 64 + n];
        bp[2 * e + 1] = (_Float16)Wd[(k0 + 1) * 64 + n];
    }
}

// -------------------------------------------------- bilinear warp primitives
__device__ __forceinline__ void warp_coords(const float* __restrict__ M,
                                            float x, float y,
                                            float& wx, float& wy) {
    float wz = M[6] * x + M[7] * y + M[8];
    wx = (M[0] * x + M[1] * y + M[2]) / wz;
    wy = (M[3] * x + M[4] * y + M[5]) / wz;
}

__device__ __forceinline__ void bilin_setup(float wx, float wy,
                                            unsigned& i00, unsigned& i01,
                                            unsigned& i10, unsigned& i11,
                                            float& w00, float& w01, float& w10, float& w11) {
    float x0 = floorf(wx), x1 = x0 + 1.0f;
    float y0 = floorf(wy), y1 = y0 + 1.0f;
    float x0s = fminf(fmaxf(x0, 0.0f), (float)(IMG_W - 1));
    float x1s = fminf(fmaxf(x1, 0.0f), (float)(IMG_W - 1));
    float y0s = fminf(fmaxf(y0, 0.0f), (float)(IMG_H - 1));
    float y1s = fminf(fmaxf(y1, 0.0f), (float)(IMG_H - 1));
    float dx = wx - x0s, dy = wy - y0s;
    float d1x = x1s - wx, d1y = y1s - wy;
    i00 = (unsigned)(int)(x0s + y0s * (float)IMG_W);
    i01 = (unsigned)(int)(x0s + y1s * (float)IMG_W);
    i10 = (unsigned)(int)(x1s + y0s * (float)IMG_W);
    i11 = (unsigned)(int)(x1s + y1s * (float)IMG_W);
    w00 = d1x * d1y; w01 = d1x * dy; w10 = dx * d1y; w11 = dx * dy;
}

// ------------------------------------------- K1: warp image_rep -> warped[320]
__global__ __launch_bounds__(256)
void warp_image_kernel(const float* __restrict__ image,     // [32,HW]
                       const float* __restrict__ invh,      // [320,9]
                       float* __restrict__ out) {           // [320,HW]
    unsigned gid = blockIdx.x * 256u + threadIdx.x;          // < 24,576,000
    if (gid >= (unsigned)NBS * HW) return;
    unsigned n = gid / HW;
    unsigned p = gid % HW;
    unsigned b = n % NB;     // jnp.tile stacking: image_rep[n] = image[n % 32]
    float x = (float)(p % IMG_W), y = (float)(p / IMG_W);
    float wx, wy;
    warp_coords(invh + n * 9, x, y, wx, wy);
    unsigned i00, i01, i10, i11; float w00, w01, w10, w11;
    bilin_setup(wx, wy, i00, i01, i10, i11, w00, w01, w10, w11);
    unsigned base = b * (unsigned)HW;
    out[gid] = image[base + i00] * w00 + image[base + i01] * w01 +
               image[base + i10] * w10 + image[base + i11] * w11;
}

// --------- K2: in-place per-8x8-block 64x64 GEMM via v_wmma_f32_16x16x32_f16
// Row r of the implicit A matrix = flattened 8x8 tile; D[r, d] maps back to
// the SAME tile, so in-place is safe (A fully in VGPRs before any store; each
// wave owns its 16 blocks exclusively). All addressing: uniform base + u32.
__global__ __launch_bounds__(256)
void gemm_wmma_kernel(float* __restrict__ wp,               // [320,HW] in-place
                      const _Float16* __restrict__ bpack) { // packed B frags
    int wave = (int)((blockIdx.x * blockDim.x + threadIdx.x) >> 5); // 0..23999
    int lane = threadIdx.x & 31;
    int mlane = lane & 15;
    int half  = (lane >> 4) & 1;

    // A tile: rows wave*16 .. wave*16+15, K = 0..63 (two K=32 chunks)
    int r   = wave * 16 + mlane;
    int img = r / 1200;
    int blk = r % 1200;
    unsigned baseA = (unsigned)img * HW + (unsigned)((blk / 40) * 8) * IMG_W
                   + (unsigned)((blk % 40) * 8);

    v16h a0, a1;
    #pragma unroll
    for (int v = 0; v < 8; ++v) {
        int grp = v >> 2, sub = v & 3;
        int k0 = grp * 16 + half * 8 + sub * 2;        // ISA 16-bit A 16x32 layout
        {
            unsigned off = baseA + (unsigned)((k0 >> 3) * IMG_W + (k0 & 7));
            float2 p = *(const float2*)(wp + off);     // pair in same tile row
            a0[2 * v]     = (_Float16)p.x;
            a0[2 * v + 1] = (_Float16)p.y;
        }
        {
            int k1 = k0 + 32;
            unsigned off = baseA + (unsigned)((k1 >> 3) * IMG_W + (k1 & 7));
            float2 q = *(const float2*)(wp + off);
            a1[2 * v]     = (_Float16)q.x;
            a1[2 * v + 1] = (_Float16)q.y;
        }
    }

    const v16h* Bp = (const v16h*)bpack;               // 32B-aligned fragments

    #pragma unroll
    for (int nt = 0; nt < 4; ++nt) {
        int n = nt * 16 + mlane;                       // output channel d
        v16h b0 = Bp[(unsigned)((nt * 2 + 0) * 32 + lane)];
        v16h b1 = Bp[(unsigned)((nt * 2 + 1) * 32 + lane)];
        v8f c = {};
        c = __builtin_amdgcn_wmma_f32_16x16x32_f16(false, a0, false, b0,
                                                   (short)0, c, false, false);
        c = __builtin_amdgcn_wmma_f32_16x16x32_f16(false, a1, false, b1,
                                                   (short)0, c, false, false);
        int dy = n >> 3, dx = n & 7;                   // pixel_shuffle8 inverse
        #pragma unroll
        for (int i = 0; i < 8; ++i) {
            int m  = i + half * 8;                     // C/D layout: vgpr i, lane half
            int rr = wave * 16 + m;
            int bi = rr / 1200, bb = rr % 1200;
            unsigned off = (unsigned)bi * HW
                         + (unsigned)((bb / 40) * 8 + dy) * IMG_W
                         + (unsigned)((bb % 40) * 8 + dx);
            wp[off] = c[i];
        }
    }
}

// -------- K3: second warp (gather warped_prob) + analytic count, sum over S
__global__ __launch_bounds__(256)
void accum_kernel(const float* __restrict__ wp,     // [320,HW] (now = prob)
                  const float* __restrict__ inv2,   // [320,9] = inv(sampled_inv_homo)
                  float* __restrict__ finalp) {     // [32,HW]
    unsigned gid = blockIdx.x * 256u + threadIdx.x;
    if (gid >= (unsigned)NB * HW) return;
    unsigned b = gid / HW;
    unsigned p = gid % HW;
    float x = (float)(p % IMG_W), y = (float)(p / IMG_W);
    float sp = 0.0f, sc = 0.0f;
    #pragma unroll
    for (int s = 0; s < NS; ++s) {
        unsigned n = (unsigned)s * NB + b;
        float wx, wy;
        warp_coords(inv2 + n * 9, x, y, wx, wy);
        unsigned i00, i01, i10, i11; float w00, w01, w10, w11;
        bilin_setup(wx, wy, i00, i01, i10, i11, w00, w01, w10, w11);
        unsigned base = n * (unsigned)HW;
        sp += wp[base + i00] * w00 + wp[base + i01] * w01 +
              wp[base + i10] * w10 + wp[base + i11] * w11;
        sc += w00 + w01 + w10 + w11;   // warp of all-ones image
    }
    finalp[gid] = sp / sc;
}

// ------------------------------------------------- K4: 9x9 spatial NMS (SAME)
__global__ __launch_bounds__(256)
void nms_kernel(const float* __restrict__ fp, float* __restrict__ out) {
    unsigned gid = blockIdx.x * 256u + threadIdx.x;
    if (gid >= (unsigned)NB * HW) return;
    unsigned p = gid % HW;
    int x = (int)(p % IMG_W), y = (int)(p / IMG_W);
    unsigned base = gid - p;                 // b*HW
    float v = fp[gid];
    int y0 = max(y - 4, 0), y1 = min(y + 4, IMG_H - 1);
    int x0 = max(x - 4, 0), x1 = min(x + 4, IMG_W - 1);
    float mx = -3.402823466e38f;
    for (int yy = y0; yy <= y1; ++yy)
        for (int xx = x0; xx <= x1; ++xx)
            mx = fmaxf(mx, fp[base + (unsigned)(yy * IMG_W + xx)]);
    out[gid] = (v == mx) ? v : 0.0f;
}

// -------- K5: per-image top-512 with reference tie rules (one block / image)
// key = (float_bits << 20) | (0xFFFFF - idx): descending sort == value desc,
// index asc. Zero-filled tail = zero-positions in ascending index order.
__global__ __launch_bounds__(512)
void topk_kernel(const float* __restrict__ nms,
                 const float* __restrict__ det_thr_p,
                 float* __restrict__ point,     // [32,512,3]
                 float* __restrict__ mask) {    // [32,512]
    __shared__ unsigned long long keys[CAND_MAX];
    __shared__ int chunkPref[512];
    __shared__ int s_cnt;
    int b   = blockIdx.x;
    int tid = threadIdx.x;
    const float* f = nms + (size_t)b * HW;
    float dt = det_thr_p[0];

    if (tid == 0) s_cnt = 0;
    for (int i = tid; i < CAND_MAX; i += 512) keys[i] = 0ULL;
    __syncthreads();

    // collect candidates (>= detection threshold); composite keys are unique
    for (int i = tid; i < HW; i += 512) {
        float v = f[i];
        if (v >= dt) {
            int pos = atomicAdd(&s_cnt, 1);
            if (pos < CAND_MAX)
                keys[pos] = ((unsigned long long)__float_as_uint(v) << 20) |
                            (unsigned long long)(0xFFFFF - i);
        }
    }
    __syncthreads();
    int cnt = s_cnt;
    int nc  = (cnt < CAND_MAX) ? cnt : CAND_MAX;

    // bitonic sort, descending (padding keys == 0 sink to the end)
    for (int k = 2; k <= CAND_MAX; k <<= 1) {
        for (int j = k >> 1; j > 0; j >>= 1) {
            for (int i = tid; i < CAND_MAX; i += 512) {
                int ixj = i ^ j;
                if (ixj > i) {
                    unsigned long long ka = keys[i], kb = keys[ixj];
                    bool descBlk = ((i & k) == 0);
                    if (descBlk ? (ka < kb) : (ka > kb)) {
                        keys[i] = kb; keys[ixj] = ka;
                    }
                }
            }
            __syncthreads();
        }
    }

    int nk = (nc < TOPK) ? nc : TOPK;
    float thr = (nc > TOPK)
        ? __uint_as_float((unsigned)(keys[TOPK - 1] >> 20))
        : dt;   // when <=512 candidates, the surviving set is exactly f >= dt

    // nonzero output slots
    for (int j = tid; j < nk; j += 512) {
        unsigned long long kk = keys[j];
        int idx = 0xFFFFF - (int)(kk & 0xFFFFFULL);
        float px = (float)(idx % IMG_W);
        float py = (float)(idx / IMG_W);
        size_t o = ((size_t)b * TOPK + j) * 3;
        point[o + 0] = px; point[o + 1] = py; point[o + 2] = 1.0f;
        mask[(size_t)b * TOPK + j] = 1.0f;
    }

    // zero-fill tail: zero positions (f < thr after thresholding) idx-ascending
    int need = TOPK - nk;
    int base = tid * 150;                       // 76800 / 512 = 150
    int zc = 0;
    for (int t = 0; t < 150; ++t) {
        float v = f[base + t];
        if (!(v >= thr)) zc++;                  // NaN counts as zero (matches jnp)
    }
    chunkPref[tid] = zc;
    __syncthreads();
    for (int off = 1; off < 512; off <<= 1) {   // inclusive Hillis-Steele scan
        int v = chunkPref[tid];
        if (tid >= off) v += chunkPref[tid - off];
        __syncthreads();
        chunkPref[tid] = v;
        __syncthreads();
    }
    if (need > 0) {
        int excl = chunkPref[tid] - zc;
        if (excl < need) {
            int rank = excl;
            for (int t = 0; t < 150 && rank < need; ++t) {
                float v = f[base + t];
                if (!(v >= thr)) {
                    int idx = base + t;
                    int j = nk + rank;
                    size_t o = ((size_t)b * TOPK + j) * 3;
                    point[o + 0] = (float)(idx % IMG_W);
                    point[o + 1] = (float)(idx / IMG_W);
                    point[o + 2] = 1.0f;
                    mask[(size_t)b * TOPK + j] = 0.0f;
                    rank++;
                }
            }
        }
    }
}

// --------------------------------------------------------------------- launch
extern "C" void kernel_launch(void* const* d_in, const int* in_sizes, int n_in,
                              void* d_out, int out_size, void* d_ws, size_t ws_size,
                              hipStream_t stream) {
    const float* image = (const float*)d_in[0];   // [32,1,240,320]
    const float* homo  = (const float*)d_in[1];   // [320,3,3]
    const float* ihomo = (const float*)d_in[2];   // [320,3,3]
    const float* Wd    = (const float*)d_in[3];   // [64,64]
    const float* det   = (const float*)d_in[4];   // scalar

    float* out = (float*)d_out;
    float* point_out = out + (size_t)NB * HW;                 // 2457600
    float* mask_out  = point_out + (size_t)NB * TOPK * 3;     // +49152

    float* ws     = (float*)d_ws;
    float* inv1   = ws;                          // 320*9
    float* inv2   = ws + 2880;                   // 320*9
    float* warped = ws + 5760;                   // 320*76800 (reused for prob)
    float* finalp = warped + (size_t)NBS * HW;   // 32*76800
    float* nmsb   = finalp + (size_t)NB * HW;    // 32*76800
    _Float16* bpack = (_Float16*)(nmsb + (size_t)NB * HW);    // 4096 halves

    // org_image passthrough
    hipMemcpyAsync(out, image, (size_t)NB * HW * sizeof(float),
                   hipMemcpyDeviceToDevice, stream);

    invert3x3_kernel<<<dim3(5), dim3(64), 0, stream>>>(homo, inv1, NBS);
    invert3x3_kernel<<<dim3(5), dim3(64), 0, stream>>>(ihomo, inv2, NBS);
    pack_b_kernel<<<dim3(1), dim3(256), 0, stream>>>(Wd, bpack);

    warp_image_kernel<<<dim3(((unsigned)NBS * HW + 255) / 256), dim3(256), 0, stream>>>(
        image, inv1, warped);

    // 24000 row-tiles of 16, 8 waves per block
    gemm_wmma_kernel<<<dim3(3000), dim3(256), 0, stream>>>(warped, bpack);

    accum_kernel<<<dim3(((unsigned)NB * HW + 255) / 256), dim3(256), 0, stream>>>(
        warped, inv2, finalp);

    nms_kernel<<<dim3(((unsigned)NB * HW + 255) / 256), dim3(256), 0, stream>>>(
        finalp, nmsb);

    topk_kernel<<<dim3(NB), dim3(512), 0, stream>>>(nmsb, det, point_out, mask_out);
}
